// SequenceRNNDecoder_12953621364951
// MI455X (gfx1250) — matrix operand order repaired and checked by reading, not compiled, not run on hardware
//
#include <hip/hip_runtime.h>
#include <hip/hip_bf16.h>
#include <math.h>

#define B_    16
#define T_    256
#define H_    512
#define V_    32000
#define G3H   1536
#define START_TOK 1

// logits B-panel staging: 64 rows of E (bf16), row stride padded 512+8
#define LROWS   64
#define LSTRIDE 520   // elements; 1040B -> 4-dword bank rotation per row

typedef __attribute__((ext_vector_type(16))) __bf16       v16bf;
typedef __attribute__((ext_vector_type(8)))  float        v8f;
typedef __attribute__((ext_vector_type(4)))  unsigned int u32x4;
typedef __attribute__((ext_vector_type(8)))  int          i32x8;
typedef __attribute__((ext_vector_type(4)))  int          i32x4;

union FragU { u32x4 u[2]; v16bf v; };

// ---- WMMA operand loaders (CDNA5 16x16x32 bf16 layouts) -------------------
// A (MxK = 16x32): lane&15 = row M; lanes 0-15 hold K {0-7,16-23}, lanes 16-31
// hold K {8-15,24-31}: two 16B chunks at koff and koff+16.
__device__ __forceinline__ v16bf load_frag_A(const unsigned short* src, int stride,
                                             int rowbase, int kbase, int lane) {
    const unsigned short* p = src + (size_t)(rowbase + (lane & 15)) * stride
                                  + kbase + ((lane & 16) ? 8 : 0);
    FragU f;
    f.u[0] = *(const u32x4*)(p);
    f.u[1] = *(const u32x4*)(p + 16);
    return f.v;
}

// B (KxN = 32x16) sourced from row-major W[n][k]: lane&15 = column N (row of W),
// lanes 0-15 hold K {0-7,16-23}, lanes 16-31 hold K {8-15,24-31} (matches A).
__device__ __forceinline__ v16bf load_frag_B(const unsigned short* w, int stride,
                                             int nbase, int kbase, int lane) {
    const unsigned short* p = w + (size_t)(nbase + (lane & 15)) * stride
                                + kbase + ((lane & 16) ? 8 : 0);
    FragU f;
    f.u[0] = *(const u32x4*)(p);
    f.u[1] = *(const u32x4*)(p + 16);
    return f.v;
}

__device__ __forceinline__ unsigned short f32_to_bf16_rne(float x) {
    unsigned int u = __float_as_uint(x);
    return (unsigned short)((u + 0x7FFFu + ((u >> 16) & 1u)) >> 16);
}

// ---- Kernel A: fp32 -> bf16 conversion ------------------------------------
__global__ void cvt_bf16_kernel(const float* __restrict__ in,
                                unsigned short* __restrict__ out, int n) {
    int i = blockIdx.x * blockDim.x + threadIdx.x;
    int stride = gridDim.x * blockDim.x;
    for (; i < n; i += stride) out[i] = f32_to_bf16_rne(in[i]);
}

// ---- Kernel B: GI[t*B+b][3H] = E_bf[tok(t,b)] @ W_ih^T + b_ih -------------
__global__ __launch_bounds__(256) void gi_kernel(
    const unsigned short* __restrict__ E_bf,
    const unsigned short* __restrict__ Wih_bf,
    const int* __restrict__ target,
    const float* __restrict__ b_ih,
    float* __restrict__ GI) {
    int lane  = threadIdx.x & 31;
    int wave  = threadIdx.x >> 5;
    int mtile = blockIdx.y * 8 + wave;   // == t, 0..255
    int ngrp  = blockIdx.x;              // 0..23 (4 N-tiles each)
    int t = mtile, b = lane & 15;
    int tok = (t == 0) ? START_TOK : target[b * T_ + (t - 1)];
    const unsigned short* arow = E_bf + (size_t)tok * H_ + ((lane & 16) ? 8 : 0);

    v8f acc[4] = {};
    for (int ks = 0; ks < H_; ks += 32) {
        FragU fa;
        fa.u[0] = *(const u32x4*)(arow + ks);
        fa.u[1] = *(const u32x4*)(arow + ks + 16);
#pragma unroll
        for (int j = 0; j < 4; ++j) {
            v16bf bf = load_frag_B(Wih_bf, H_, (ngrp * 4 + j) * 16, ks, lane);
            acc[j] = __builtin_amdgcn_wmma_f32_16x16x32_bf16(
                false, fa.v, false, bf, (short)0, acc[j], false, false);
        }
    }
    int mloc = mtile * 16 + ((lane & 16) ? 8 : 0);
#pragma unroll
    for (int j = 0; j < 4; ++j) {
        int n = (ngrp * 4 + j) * 16 + (lane & 15);
        float bias = b_ih[n];
#pragma unroll
        for (int v = 0; v < 8; ++v)
            GI[(size_t)(mloc + v) * G3H + n] = acc[j][v] + bias;
    }
}

// ---- Kernel C: sequential GRU, single workgroup, 16 waves -----------------
__global__ __launch_bounds__(512) void gru_seq_kernel(
    const float* __restrict__ hidden0,
    const float* __restrict__ GI,
    const unsigned short* __restrict__ Whh_bf,
    const float* __restrict__ b_hh,
    unsigned short* __restrict__ Hseq_bf) {
    __shared__ __align__(16) float          h_cur[B_][H_];
    __shared__ __align__(16) unsigned short h_bf[B_][H_];
    int tid = threadIdx.x, lane = tid & 31, wave = tid >> 5;

    float* hc = &h_cur[0][0];
    unsigned short* hb = &h_bf[0][0];
    for (int i = tid; i < B_ * H_; i += 512) {
        float v = hidden0[i];
        hc[i] = v;
        hb[i] = f32_to_bf16_rne(v);
    }
    int nloc = lane & 15, bbase = (lane & 16) ? 8 : 0;
    float bhr[2], bhz[2], bhn[2];
#pragma unroll
    for (int p = 0; p < 2; ++p) {
        int j = wave * 32 + p * 16 + nloc;
        bhr[p] = b_hh[j];
        bhz[p] = b_hh[H_ + j];
        bhn[p] = b_hh[2 * H_ + j];
    }
    __syncthreads();

    for (int t = 0; t < T_; ++t) {
        v8f ar[2] = {}, az[2] = {}, an[2] = {};
        for (int ks = 0; ks < H_; ks += 32) {
            v16bf a = load_frag_A(hb, H_, 0, ks, lane);
#pragma unroll
            for (int p = 0; p < 2; ++p) {
                int nt = wave * 2 + p;
                v16bf wr = load_frag_B(Whh_bf, H_, nt * 16, ks, lane);
                v16bf wz = load_frag_B(Whh_bf, H_, (32 + nt) * 16, ks, lane);
                v16bf wn = load_frag_B(Whh_bf, H_, (64 + nt) * 16, ks, lane);
                ar[p] = __builtin_amdgcn_wmma_f32_16x16x32_bf16(
                    false, a, false, wr, (short)0, ar[p], false, false);
                az[p] = __builtin_amdgcn_wmma_f32_16x16x32_bf16(
                    false, a, false, wz, (short)0, az[p], false, false);
                an[p] = __builtin_amdgcn_wmma_f32_16x16x32_bf16(
                    false, a, false, wn, (short)0, an[p], false, false);
            }
        }
        __syncthreads();   // all waves done reading h_bf
#pragma unroll
        for (int p = 0; p < 2; ++p) {
            int j = wave * 32 + p * 16 + nloc;
#pragma unroll
            for (int v = 0; v < 8; ++v) {
                int b = bbase + v;
                size_t g = ((size_t)t * B_ + b) * G3H;
                float gir = GI[g + j];
                float giz = GI[g + H_ + j];
                float gin = GI[g + 2 * H_ + j];
                float ghr = ar[p][v] + bhr[p];
                float ghz = az[p][v] + bhz[p];
                float ghn = an[p][v] + bhn[p];
                float r = 1.0f / (1.0f + expf(-(gir + ghr)));
                float z = 1.0f / (1.0f + expf(-(giz + ghz)));
                float n = tanhf(gin + r * ghn);
                float hn = (1.0f - z) * n + z * h_cur[b][j];
                h_cur[b][j] = hn;
                unsigned short h16 = f32_to_bf16_rne(hn);
                h_bf[b][j] = h16;
                Hseq_bf[((size_t)b * T_ + t) * H_ + j] = h16;  // [b][t][h]
            }
        }
        __syncthreads();
    }
}

// ---- Kernel D: out[b][t][v] = Hseq[b][t][:] @ E^T + b_out -----------------
// Block = 8 waves sharing one 64-column E panel staged in LDS by the TDM.
// D# per ISA ch.8: 2D tile, data_size=2B, tile 512x64, pad 4 dwords per
// 256 dwords -> LDS row stride 520 elements (bank-conflict-free frag reads).
__global__ __launch_bounds__(256) void logits_kernel(
    const unsigned short* __restrict__ Hseq_bf,
    const unsigned short* __restrict__ E_bf,
    const float* __restrict__ b_out,
    float* __restrict__ out) {
    __shared__ __align__(16) unsigned short etile[LROWS * LSTRIDE]; // 66,560 B
    int lane  = threadIdx.x & 31;
    int wave  = threadIdx.x >> 5;
    int mtile = blockIdx.y * 8 + wave;  // 0..255
    int ngrp  = blockIdx.x;             // 0..499 -> columns [ngrp*64, +64)
    size_t nbase = (size_t)ngrp * 64;

#if __has_builtin(__builtin_amdgcn_tensor_load_to_lds)
    if (wave == 0) {
        unsigned long long gaddr =
            (unsigned long long)(const void*)(E_bf + nbase * H_);
        unsigned int lds_addr =
            (unsigned int)(unsigned long long)(const void*)&etile[0];
        // D# group 0: count=1 | lds_addr | global_addr[56:0] | type=2
        u32x4 g0 = { 1u, lds_addr, (unsigned int)gaddr,
                     ((unsigned int)(gaddr >> 32) & 0x01FFFFFFu) | (2u << 30) };
        // D# group 1: data_size=1(2B), pad_enable, pad_interval=7(256dw),
        // pad_amount=3(4dw=16B); tensor_dim0=512, tensor_dim1=64,
        // tile_dim0=512, tile_dim1=64, tensor_dim0_stride=512
        i32x8 g1 = { (int)((1u << 16) | (1u << 20) | (7u << 22) | (3u << 25)),
                     (int)(512u << 16),   // tensor_dim0[15:0] << 16
                     (int)(64u  << 16),   // tensor_dim1[15:0] << 16
                     (int)(512u << 16),   // tile_dim0 << 16
                     64,                  // tile_dim1
                     512,                 // tensor_dim0_stride[31:0]
                     0, 0 };
        i32x4 gz4 = { 0, 0, 0, 0 };
        i32x8 gz8 = { 0, 0, 0, 0, 0, 0, 0, 0 };
        __builtin_amdgcn_tensor_load_to_lds(g0, g1, gz4, gz4, gz8, 0);
        __builtin_amdgcn_s_wait_tensorcnt((short)0);
    }
    __syncthreads();
#else
    // Cooperative fallback copy (also used by the host compilation pass)
    for (int i = threadIdx.x; i < LROWS * (H_ / 8); i += 256) {
        int r = i / (H_ / 8), c = (i % (H_ / 8)) * 8;
        *(u32x4*)&etile[r * LSTRIDE + c] =
            *(const u32x4*)(E_bf + (nbase + r) * H_ + c);
    }
    __syncthreads();
#endif

    const unsigned short* arow =
        Hseq_bf + (size_t)(mtile * 16 + (lane & 15)) * H_ + ((lane & 16) ? 8 : 0);
    int koffB = (lane & 16) ? 8 : 0;

    v8f acc[4] = {};
    for (int ks = 0; ks < H_; ks += 32) {
        FragU fa;
        fa.u[0] = *(const u32x4*)(arow + ks);
        fa.u[1] = *(const u32x4*)(arow + ks + 16);
#pragma unroll
        for (int j = 0; j < 4; ++j) {
            const unsigned short* p =
                &etile[(j * 16 + (lane & 15)) * LSTRIDE + ks + koffB];
            FragU fb;
            fb.u[0] = *(const u32x4*)(p);
            fb.u[1] = *(const u32x4*)(p + 16);
            acc[j] = __builtin_amdgcn_wmma_f32_16x16x32_bf16(
                false, fa.v, false, fb.v, (short)0, acc[j], false, false);
        }
    }
    int mloc = mtile * 16 + ((lane & 16) ? 8 : 0);
#pragma unroll
    for (int j = 0; j < 4; ++j) {
        int n = (int)nbase + j * 16 + (lane & 15);
        float bias = b_out[n];
#pragma unroll
        for (int v = 0; v < 8; ++v)
            out[(size_t)(mloc + v) * V_ + n] = acc[j][v] + bias;
    }
}

// ---- Host launch ----------------------------------------------------------
extern "C" void kernel_launch(void* const* d_in, const int* in_sizes, int n_in,
                              void* d_out, int out_size, void* d_ws, size_t ws_size,
                              hipStream_t stream) {
    const float* hidden0 = (const float*)d_in[0];
    const int*   target  = (const int*)d_in[1];
    const float* E       = (const float*)d_in[2];
    const float* W_ih    = (const float*)d_in[3];
    const float* W_hh    = (const float*)d_in[4];
    const float* b_ih    = (const float*)d_in[5];
    const float* b_hh    = (const float*)d_in[6];
    const float* b_out   = (const float*)d_in[7];
    float* out = (float*)d_out;

    // Workspace layout (all offsets 16B aligned), total ~65.3 MB
    char* ws = (char*)d_ws;
    unsigned short* E_bf    = (unsigned short*)(ws);                    // 32,768,000 B
    unsigned short* Wih_bf  = (unsigned short*)(ws + 32768000);         //  1,572,864 B
    unsigned short* Whh_bf  = (unsigned short*)(ws + 32768000 + 1572864);
    float*          GI      = (float*)(ws + 32768000 + 2 * 1572864);    // 25,165,824 B
    unsigned short* Hseq_bf = (unsigned short*)(ws + 32768000 + 2 * 1572864 + 25165824);

    // 1) fp32 -> bf16 operand conversion (E is L2-resident at 32 MB)
    cvt_bf16_kernel<<<dim3(2048), 256, 0, stream>>>(E, E_bf, V_ * H_);
    cvt_bf16_kernel<<<dim3(256), 256, 0, stream>>>(W_ih, Wih_bf, G3H * H_);
    cvt_bf16_kernel<<<dim3(256), 256, 0, stream>>>(W_hh, Whh_bf, G3H * H_);

    // 2) all input-side gate pre-activations in one parallel WMMA GEMM
    gi_kernel<<<dim3(24, 32), 256, 0, stream>>>(E_bf, Wih_bf, target, b_ih, GI);

    // 3) serial GRU recurrence (single workgroup, WMMA per step)
    gru_seq_kernel<<<dim3(1), 512, 0, stream>>>(hidden0, GI, Whh_bf, b_hh, Hseq_bf);

    // 4) tied-embedding logits: 4096 x 32000 x 512 WMMA GEMM, TDM-staged B
    logits_kernel<<<dim3(500, 32), 256, 0, stream>>>(Hseq_bf, E_bf, b_out, out);

    (void)in_sizes; (void)n_in; (void)out_size; (void)ws_size;
}
